// DecoderGrp_30382598652304
// MI455X (gfx1250) — matrix operand order repaired and checked by reading, not compiled, not run on hardware
//
#include <hip/hip_runtime.h>
#include <hip/hip_bf16.h>
#include <cstdint>

typedef __attribute__((ext_vector_type(16))) _Float16 v16h;
typedef __attribute__((ext_vector_type(8)))  float    v8f;

#define NPTS   8192
#define MPTS   2048
#define KNN    16
#define INDIM  128
#define FDIM   131
#define FPAD   160
#define DIMC   256
#define NBATCH 8

// ---------------------------------------------------------------------------
// Kernel 0: pre-swizzle W1 (131x256, pad K->160) and W2 (256x256) into f16
// WMMA B-fragment order: frag f=(kc*16+nt), within frag element [lane][h],
// value = W[kc*32 + (h&7) + 8*(lane/16) + 16*(h/8)][nt*16 + lane%16]
// ---------------------------------------------------------------------------
__global__ void preswz_kernel(const float* __restrict__ W1,
                              const float* __restrict__ W2,
                              _Float16* __restrict__ W1p,
                              _Float16* __restrict__ W2p) {
    int tid = blockIdx.x * blockDim.x + threadIdx.x;
    const int n1 = 5 * 16 * 512;   // W1 frags: 5 K-chunks x 16 N-tiles
    const int n2 = 8 * 16 * 512;   // W2 frags: 8 K-chunks x 16 N-tiles
    if (tid < n1) {
        int f = tid / 512, r = tid % 512;
        int l = r / 16, h = r % 16;
        int kc = f / 16, nt = f % 16;
        int K = kc * 32 + (h & 7) + ((l >> 4) << 3) + ((h >> 3) << 4);
        int N = nt * 16 + (l & 15);
        float v = (K < FDIM) ? W1[K * DIMC + N] : 0.0f;
        W1p[tid] = (_Float16)v;
    } else if (tid < n1 + n2) {
        int t2 = tid - n1;
        int f = t2 / 512, r = t2 % 512;
        int l = r / 16, h = r % 16;
        int kc = f / 16, nt = f % 16;
        int K = kc * 32 + (h & 7) + ((l >> 4) << 3) + ((h >> 3) << 4);
        int N = nt * 16 + (l & 15);
        W2p[t2] = (_Float16)W2[K * DIMC + N];
    }
}

// ---------------------------------------------------------------------------
// Kernel 1: farthest point sampling. One 1024-thread block per batch.
// Point coords live in registers (8 points/thread); per-step block argmax.
// ---------------------------------------------------------------------------
__global__ __launch_bounds__(1024) void fps_kernel(const float* __restrict__ dec_pc,
                                                   float* __restrict__ fps_out) {
    int b = blockIdx.x;
    int t = threadIdx.x;
    const float* pcb = dec_pc + (size_t)b * NPTS * 3;

    float px[8], py[8], pz[8], md[8];
#pragma unroll
    for (int j = 0; j < 8; ++j) {
        int p = t + j * 1024;
        px[j] = pcb[p * 3 + 0];
        py[j] = pcb[p * 3 + 1];
        pz[j] = pcb[p * 3 + 2];
        md[j] = 3.4e38f;
    }

    __shared__ float rv[32];
    __shared__ int   ri[32];
    __shared__ float selc[3];
    __shared__ int   selIdx;
    if (t == 0) { selc[0] = px[0]; selc[1] = py[0]; selc[2] = pz[0]; selIdx = 0; }
    __syncthreads();

    for (int s = 0; s < MPTS; ++s) {
        float qx = selc[0], qy = selc[1], qz = selc[2];
        if (t == 0) {
            size_t o = ((size_t)b * MPTS + s) * 3;
            fps_out[o + 0] = qx; fps_out[o + 1] = qy; fps_out[o + 2] = qz;
        }
        float bv = -1.0f; int bi = 0;
#pragma unroll
        for (int j = 0; j < 8; ++j) {
            float dx = px[j] - qx, dy = py[j] - qy, dz = pz[j] - qz;
            float d = dx * dx + dy * dy + dz * dz;
            md[j] = fminf(md[j], d);
            if (md[j] > bv) { bv = md[j]; bi = t + j * 1024; }
        }
        // wave32 reduction (max val, min idx on tie)
#pragma unroll
        for (int off = 16; off > 0; off >>= 1) {
            float ov = __shfl_down(bv, off, 32);
            int   oi = __shfl_down(bi, off, 32);
            if (ov > bv || (ov == bv && oi < bi)) { bv = ov; bi = oi; }
        }
        int wid = t >> 5;
        if ((t & 31) == 0) { rv[wid] = bv; ri[wid] = bi; }
        __syncthreads();
        if (t < 32) {
            bv = rv[t]; bi = ri[t];
#pragma unroll
            for (int off = 16; off > 0; off >>= 1) {
                float ov = __shfl_down(bv, off, 32);
                int   oi = __shfl_down(bi, off, 32);
                if (ov > bv || (ov == bv && oi < bi)) { bv = ov; bi = oi; }
            }
            if (t == 0) selIdx = bi;
        }
        __syncthreads();
        int w = selIdx;
        if ((w & 1023) == t) {          // owner publishes coords for next step
            int jj = w >> 10;
            float cx = px[0], cy = py[0], cz = pz[0];
#pragma unroll
            for (int j = 1; j < 8; ++j)
                if (jj == j) { cx = px[j]; cy = py[j]; cz = pz[j]; }
            selc[0] = cx; selc[1] = cy; selc[2] = cz;
        }
        __syncthreads();
    }
}

// ---------------------------------------------------------------------------
// Kernel 2: 16-NN (unordered; max-pool later is order invariant).
// One thread per query point, register-resident replace-max top-16 list.
// ---------------------------------------------------------------------------
#define TS 1024
__global__ __launch_bounds__(128) void knn_kernel(const float* __restrict__ dec_pc,
                                                  const float* __restrict__ fps_pc,
                                                  int* __restrict__ knn_idx) {
    __shared__ float tx[TS], ty[TS], tz[TS];
    int b = blockIdx.y;
    int m = blockIdx.x * 128 + threadIdx.x;
    int bm = b * MPTS + m;
    const float* pcb = dec_pc + (size_t)b * NPTS * 3;
    float qx = fps_pc[bm * 3 + 0];
    float qy = fps_pc[bm * 3 + 1];
    float qz = fps_pc[bm * 3 + 2];

    float knd[KNN]; int kni[KNN];
#pragma unroll
    for (int i = 0; i < KNN; ++i) { knd[i] = 3.4e38f; kni[i] = 0; }
    float dmax = 3.4e38f;

    for (int t0 = 0; t0 < NPTS; t0 += TS) {
        __syncthreads();
        for (int e = threadIdx.x; e < TS; e += 128) {
            tx[e] = pcb[(t0 + e) * 3 + 0];
            ty[e] = pcb[(t0 + e) * 3 + 1];
            tz[e] = pcb[(t0 + e) * 3 + 2];
        }
        __syncthreads();
        for (int c = 0; c < TS; ++c) {
            float dx = tx[c] - qx, dy = ty[c] - qy, dz = tz[c] - qz;
            float d = dx * dx + dy * dy + dz * dz;
            if (d < dmax) {
                bool done = false;
#pragma unroll
                for (int i = 0; i < KNN; ++i)
                    if (!done && knd[i] == dmax) { knd[i] = d; kni[i] = t0 + c; done = true; }
                dmax = knd[0];
#pragma unroll
                for (int i = 1; i < KNN; ++i) dmax = fmaxf(dmax, knd[i]);
            }
        }
    }
#pragma unroll
    for (int i = 0; i < KNN; ++i) knn_idx[bm * KNN + i] = kni[i];
}

// ---------------------------------------------------------------------------
// Kernel 3: fused grouped MLP with WMMA. One block (4 waves) per (b,m).
// A tile = the 16 KNN rows of this sample. relu(A@W1+b1)@W2+b2, col-max.
// ---------------------------------------------------------------------------
__global__ __launch_bounds__(128) void mlp_kernel(const float* __restrict__ dec_x,
                                                  const float* __restrict__ dec_pc,
                                                  const float* __restrict__ b1,
                                                  const float* __restrict__ b2,
                                                  const v16h* __restrict__ W1p,
                                                  const v16h* __restrict__ W2p,
                                                  const int* __restrict__ knn_idx,
                                                  const float* __restrict__ fps_pc,
                                                  float* __restrict__ out) {
    __shared__ _Float16 Ash[16][FPAD];
    __shared__ _Float16 Hsh[16][DIMC];
    __shared__ int   sIdx[16];
    __shared__ float sQ[3];

    int bm = blockIdx.x;
    int b = bm >> 11;
    int tid = threadIdx.x;

    if (tid < 16) sIdx[tid] = knn_idx[bm * KNN + tid];
    if (tid < 3)  sQ[tid]   = fps_pc[bm * 3 + tid];
    __syncthreads();

    const float* pcb = dec_pc + (size_t)b * NPTS * 3;
    const float* xb  = dec_x  + (size_t)b * NPTS * INDIM;
    for (int e = tid; e < 16 * FPAD; e += 128) {
        int r = e / FPAD, c = e % FPAD;
        int n = sIdx[r];
        float v;
        if (c < 3)          v = pcb[n * 3 + c] - sQ[c];
        else if (c < FDIM)  v = xb[(size_t)n * INDIM + (c - 3)];
        else                v = 0.0f;
        Ash[r][c] = (_Float16)v;
    }
    __syncthreads();

    int wv   = tid >> 5;
    int lane = tid & 31;
    int row  = lane & 15;           // A row this lane provides
    int koff = (lane >> 4) << 3;    // +8 K for upper half-wave (A layout)
    int colo = lane & 15;           // C/D column for this lane

    // ---- layer 1: each wave owns 64 output columns (4 N-tiles) ----
    for (int ntl = 0; ntl < 4; ++ntl) {
        int nt = wv * 4 + ntl;
        int n0 = nt * 16;
        v8f acc = {};
        for (int kc = 0; kc < 5; ++kc) {
            v16h a;
            int kb = kc * 32 + koff;
#pragma unroll
            for (int h = 0; h < 16; ++h)
                a[h] = Ash[row][kb + (h & 7) + ((h >> 3) << 4)];
            v16h bf = W1p[(kc * 16 + nt) * 32 + lane];
            acc = __builtin_amdgcn_wmma_f32_16x16x32_f16(
                false, a, false, bf, (short)0, acc, false, false);
        }
        float bb = b1[n0 + colo];
#pragma unroll
        for (int v = 0; v < 8; ++v) {
            float hv = fmaxf(acc[v] + bb, 0.0f);
            Hsh[v + koff][n0 + colo] = (_Float16)hv;   // C layout: M = v (+8 upper lanes)
        }
    }
    __syncthreads();

    // ---- layer 2 + column max-pool over the 16 rows ----
    for (int ntl = 0; ntl < 4; ++ntl) {
        int nt = wv * 4 + ntl;
        int n0 = nt * 16;
        v8f acc = {};
        for (int kc = 0; kc < 8; ++kc) {
            v16h a;
            int kb = kc * 32 + koff;
#pragma unroll
            for (int h = 0; h < 16; ++h)
                a[h] = Hsh[row][kb + (h & 7) + ((h >> 3) << 4)];
            v16h bf = W2p[(kc * 16 + nt) * 32 + lane];
            acc = __builtin_amdgcn_wmma_f32_16x16x32_f16(
                false, a, false, bf, (short)0, acc, false, false);
        }
        float mx = acc[0];
#pragma unroll
        for (int v = 1; v < 8; ++v) mx = fmaxf(mx, acc[v]);
        mx += b2[n0 + colo];
        float o = fmaxf(mx, __shfl_xor(mx, 16, 32));   // rows 0-7 vs rows 8-15
        if (lane < 16) out[(size_t)bm * DIMC + n0 + lane] = o;
    }
}

// ---------------------------------------------------------------------------
extern "C" void kernel_launch(void* const* d_in, const int* in_sizes, int n_in,
                              void* d_out, int out_size, void* d_ws, size_t ws_size,
                              hipStream_t stream) {
    const float* dec_x  = (const float*)d_in[0];
    const float* dec_pc = (const float*)d_in[1];
    const float* enc_x  = (const float*)d_in[2];
    const float* enc_pc = (const float*)d_in[3];
    const float* W1     = (const float*)d_in[4];
    const float* b1     = (const float*)d_in[5];
    const float* W2     = (const float*)d_in[6];
    const float* b2     = (const float*)d_in[7];

    float* out        = (float*)d_out;                       // [8,2048,256]
    float* fps_out    = out + (size_t)NBATCH * MPTS * DIMC;  // [8,2048,3]
    float* enc_x_out  = fps_out + (size_t)NBATCH * MPTS * 3; // [8,512,128]
    float* enc_pc_out = enc_x_out + (size_t)NBATCH * 512 * INDIM; // [8,512,3]

    char* ws = (char*)d_ws;
    int*      knn_idx = (int*)ws;                                   // 1 MB
    _Float16* W1p     = (_Float16*)(ws + 1048576);                  // 80 KB
    _Float16* W2p     = (_Float16*)(ws + 1048576 + 81920);          // 128 KB

    // pass-through outputs
    hipMemcpyAsync(enc_x_out, enc_x, (size_t)NBATCH * 512 * INDIM * sizeof(float),
                   hipMemcpyDeviceToDevice, stream);
    hipMemcpyAsync(enc_pc_out, enc_pc, (size_t)NBATCH * 512 * 3 * sizeof(float),
                   hipMemcpyDeviceToDevice, stream);

    preswz_kernel<<<(5 * 16 * 512 + 8 * 16 * 512 + 255) / 256, 256, 0, stream>>>(
        W1, W2, W1p, W2p);
    fps_kernel<<<NBATCH, 1024, 0, stream>>>(dec_pc, fps_out);
    knn_kernel<<<dim3(MPTS / 128, NBATCH), 128, 0, stream>>>(dec_pc, fps_out, knn_idx);
    mlp_kernel<<<NBATCH * MPTS, 128, 0, stream>>>(
        dec_x, dec_pc, b1, b2, (const v16h*)W1p, (const v16h*)W2p, knn_idx, fps_out, out);
}
